// CausalSelfAttention_69243462746242
// MI455X (gfx1250) — matrix operand order repaired and checked by reading, not compiled, not run on hardware
//
#include <hip/hip_runtime.h>

typedef __attribute__((ext_vector_type(8)))  float  v8f;
typedef __attribute__((ext_vector_type(16))) __bf16 v16bf;
typedef __attribute__((ext_vector_type(8)))  __bf16 v8bf;

#define D_MODEL  1024
#define N_HEADS  16
#define HEAD_DIM 64
#define SEQ      2048
#define BATCH    4
#define M_TOK    (BATCH * SEQ)   // 8192 tokens

#if defined(__has_builtin)
#  if __has_builtin(__builtin_amdgcn_sched_group_barrier)
#    define SCHED_GROUP(mask, size) __builtin_amdgcn_sched_group_barrier((mask), (size), 0)
#  endif
#endif
#ifndef SCHED_GROUP
#  define SCHED_GROUP(mask, size)
#endif
// masks: 0x008 = MFMA/WMMA, 0x020 = VMEM read

// ---------- helpers ----------

static __device__ inline __bf16 f2bf(float f) {
    unsigned u = __builtin_bit_cast(unsigned, f);
    unsigned r = (u + 0x7FFFu + ((u >> 16) & 1u)) >> 16;   // round-to-nearest-even
    unsigned short s = (unsigned short)r;
    return __builtin_bit_cast(__bf16, s);
}

static __device__ inline v8f v8f_zero() {
    v8f z;
    #pragma unroll
    for (int i = 0; i < 8; ++i) z[i] = 0.0f;
    return z;
}

union FragBF {
    v16bf v;
    v8bf  h[2];
};

// A fragment: 16x32 bf16 of row-major [.., lda] matrix at (row, kbase).
// Lane layout (ISA 7.12.2): lanes 0-15 rows m, half selects K {0..7,16..23} vs {8..15,24..31}.
static __device__ inline v16bf load_a_frag(const __bf16* base, int lda, int row, int kbase, int lane) {
    const int m  = row + (lane & 15);
    const int hh = lane >> 4;
    const __bf16* p = base + (size_t)m * lda + kbase + hh * 8;
    FragBF f;
    f.h[0] = *(const v8bf*)p;           // K = hh*8 + 0..7
    f.h[1] = *(const v8bf*)(p + 16);    // K = 16 + hh*8 + 0..7
    return f.v;
}

// B fragment: 32x16 bf16. "bt" is B stored transposed row-major [N, ldb=K].
// Lane layout: lane%16 = n, half selects K {0..7,16..23} vs {8..15,24..31} (matches A).
static __device__ inline v16bf load_b_frag(const __bf16* bt, int ldb, int ncol, int kbase, int lane) {
    const int n  = ncol + (lane & 15);
    const int hh = lane >> 4;
    const __bf16* p = bt + (size_t)n * ldb + kbase + hh * 8;
    FragBF f;
    f.h[0] = *(const v8bf*)p;           // K = hh*8 + 0..7
    f.h[1] = *(const v8bf*)(p + 16);    // K = 16 + hh*8 + 0..7
    return f.v;
}

// ---------- conversion kernels ----------

__global__ void f32_to_bf16_kernel(const float* __restrict__ in, __bf16* __restrict__ out, int n) {
    int i = blockIdx.x * blockDim.x + threadIdx.x;
    if (i < n) out[i] = f2bf(in[i]);
}

// W: [K,N] row-major f32  ->  Wt: [N,K] row-major bf16
__global__ void transpose_bf16_kernel(const float* __restrict__ W, __bf16* __restrict__ Wt, int K, int N) {
    int i = blockIdx.x * blockDim.x + threadIdx.x;
    if (i < N * K) {
        int n = i / K;
        int k = i - n * K;
        Wt[i] = f2bf(W[(size_t)k * N + n]);
    }
}

// ---------- pipelined GEMM core (software ping-pong over k) ----------
// Computes a 16x128 strip of C per wave: acc[8] 16x16 tiles.
// Loads for step k+32 are issued while WMMAs of step k execute.

#define GEMM_CORE(Ap, Bp)                                                                 \
    v8f acc[8];                                                                           \
    _Pragma("unroll") for (int i = 0; i < 8; ++i) acc[i] = v8f_zero();                    \
    v16bf a0 = load_a_frag((Ap), D_MODEL, mrow, 0, lane);                                 \
    v16bf a1;                                                                             \
    v16bf b0[8], b1[8];                                                                   \
    _Pragma("unroll") for (int nt = 0; nt < 8; ++nt)                                      \
        b0[nt] = load_b_frag((Bp), D_MODEL, nbase + nt * 16, 0, lane);                    \
    _Pragma("unroll 1") for (int k = 0; k < D_MODEL; k += 64) {                           \
        const int k1 = k + 32;                                                            \
        a1 = load_a_frag((Ap), D_MODEL, mrow, k1, lane);                                  \
        _Pragma("unroll") for (int nt = 0; nt < 8; ++nt)                                  \
            b1[nt] = load_b_frag((Bp), D_MODEL, nbase + nt * 16, k1, lane);               \
        _Pragma("unroll") for (int nt = 0; nt < 8; ++nt)                                  \
            acc[nt] = __builtin_amdgcn_wmma_f32_16x16x32_bf16(                            \
                false, a0, false, b0[nt], (short)0, acc[nt], false, false);               \
        const int k2 = (k + 64 < D_MODEL) ? (k + 64) : 0;                                 \
        a0 = load_a_frag((Ap), D_MODEL, mrow, k2, lane);                                  \
        _Pragma("unroll") for (int nt = 0; nt < 8; ++nt)                                  \
            b0[nt] = load_b_frag((Bp), D_MODEL, nbase + nt * 16, k2, lane);               \
        _Pragma("unroll") for (int nt = 0; nt < 8; ++nt)                                  \
            acc[nt] = __builtin_amdgcn_wmma_f32_16x16x32_bf16(                            \
                false, a1, false, b1[nt], (short)0, acc[nt], false, false);               \
        SCHED_GROUP(0x020, 18);  /* 18 VMEM reads (stage 0 prefetch)   */                 \
        SCHED_GROUP(0x008, 8);   /* 8 WMMA       (stage 0 compute)    */                  \
        SCHED_GROUP(0x020, 18);  /* 18 VMEM reads (stage 1 prefetch)   */                 \
        SCHED_GROUP(0x008, 8);   /* 8 WMMA       (stage 1 compute)    */                  \
    }

// ---------- QKV projection GEMM: [8192,1024] x [1024,3072] + bias ----------
// Scatters Q,K -> [B,H,T,Hd] bf16 ; V -> [B,H,Hd,T] bf16 (transposed for PV matmul)

__global__ __launch_bounds__(256)
__attribute__((amdgpu_waves_per_eu(1, 2)))
void qkv_gemm_kernel(
    const __bf16* __restrict__ X, const __bf16* __restrict__ Wt,
    const float* __restrict__ bias,
    __bf16* __restrict__ Qb, __bf16* __restrict__ Kb, __bf16* __restrict__ Vt)
{
    const int lane  = threadIdx.x & 31;
    const int wv    = threadIdx.x >> 5;
    const int nbase = blockIdx.x * 128;
    const int mrow  = blockIdx.y * 128 + wv * 16;

    GEMM_CORE(X, Wt)

    const int rlo = lane & 15, hh = lane >> 4;
    #pragma unroll
    for (int nt = 0; nt < 8; ++nt) {
        const int ng  = nbase + nt * 16 + rlo;
        const float bia = bias[ng];
        const int i3  = ng >> 10;
        const int rem = ng & 1023;
        const int h   = rem >> 6;
        const int hd  = rem & 63;
        #pragma unroll
        for (int r = 0; r < 8; ++r) {
            const int mg = mrow + r + hh * 8;
            const int b_ = mg >> 11;
            const int t  = mg & 2047;
            const int bh = b_ * N_HEADS + h;
            __bf16 bv = f2bf(acc[nt][r] + bia);
            if (i3 == 0)      Qb[((size_t)bh * SEQ + t) * HEAD_DIM + hd] = bv;
            else if (i3 == 1) Kb[((size_t)bh * SEQ + t) * HEAD_DIM + hd] = bv;
            else              Vt[((size_t)bh * HEAD_DIM + hd) * SEQ + t] = bv;
        }
    }
}

// ---------- flash-style causal attention ----------
// grid: (B*H, T/128); block: 256 (8 waves); wave = 16 queries, streams key tiles of 32.

__global__ __launch_bounds__(256) void attn_kernel(
    const __bf16* __restrict__ Qb, const __bf16* __restrict__ Kb,
    const __bf16* __restrict__ Vt, __bf16* __restrict__ Obf)
{
    __shared__ __align__(16) __bf16 plds[8][16][32];   // per-wave P tile (C-layout -> A-layout bounce)

    const int lane  = threadIdx.x & 31;
    const int wv    = threadIdx.x >> 5;
    const int bh    = blockIdx.x;
    const int b     = bh >> 4;
    const int h     = bh & 15;
    const int qbase = blockIdx.y * 128 + wv * 16;
    const float scale = 0.125f;                        // 1/sqrt(64)

    const __bf16* Qp = Qb + (size_t)bh * SEQ * HEAD_DIM;
    const __bf16* Kp = Kb + (size_t)bh * SEQ * HEAD_DIM;
    const __bf16* Vp = Vt + (size_t)bh * HEAD_DIM * SEQ;

    const v16bf qa0 = load_a_frag(Qp, HEAD_DIM, qbase, 0,  lane);
    const v16bf qa1 = load_a_frag(Qp, HEAD_DIM, qbase, 32, lane);

    float mrun[8], lrun[8];
    v8f oacc[4];
    #pragma unroll
    for (int r = 0; r < 8; ++r) { mrun[r] = -3.0e38f; lrun[r] = 0.0f; }
    #pragma unroll
    for (int nt = 0; nt < 4; ++nt) oacc[nt] = v8f_zero();

    const int rlo = lane & 15, hh = lane >> 4;
    const int qlast = qbase + 15;

    for (int kb = 0; kb <= qlast; kb += 32) {
        // ---- batch K-fragment loads, then score WMMAs ----
        v16bf kf[4];
        #pragma unroll
        for (int j = 0; j < 2; ++j) {
            kf[j * 2 + 0] = load_b_frag(Kp, HEAD_DIM, kb + j * 16, 0,  lane);
            kf[j * 2 + 1] = load_b_frag(Kp, HEAD_DIM, kb + j * 16, 32, lane);
        }
        v8f s[2];
        #pragma unroll
        for (int j = 0; j < 2; ++j) {
            s[j] = v8f_zero();
            s[j] = __builtin_amdgcn_wmma_f32_16x16x32_bf16(
                false, qa0, false, kf[j * 2 + 0], (short)0, s[j], false, false);
            s[j] = __builtin_amdgcn_wmma_f32_16x16x32_bf16(
                false, qa1, false, kf[j * 2 + 1], (short)0, s[j], false, false);
        }

        // ---- V-fragment loads now: latency hides under softmax VALU ----
        v16bf vb[4];
        #pragma unroll
        for (int nt = 0; nt < 4; ++nt)
            vb[nt] = load_b_frag(Vp, SEQ, nt * 16, kb, lane);

        // ---- prefetch next key tile (each K row = one 128B line) ----
        const int kpf = kb + 32;
        if (kpf <= qlast) {
            __builtin_prefetch(Kp + (size_t)(kpf + rlo) * HEAD_DIM + hh * 32, 0, 3);
            __builtin_prefetch(Kp + (size_t)(kpf + 16 + rlo) * HEAD_DIM + hh * 32, 0, 3);
            #pragma unroll
            for (int nt = 0; nt < 4; ++nt)
                __builtin_prefetch(Vp + (size_t)(nt * 16 + rlo) * SEQ + kpf + hh * 16, 0, 3);
        }

        // scale + causal mask + per-row max (over this tile)
        float mx[8];
        #pragma unroll
        for (int r = 0; r < 8; ++r) {
            const int row = qbase + r + hh * 8;
            #pragma unroll
            for (int j = 0; j < 2; ++j) {
                const int col = kb + j * 16 + rlo;
                float v = s[j][r] * scale;
                if (col > row) v = -3.0e38f;
                s[j][r] = v;
            }
            mx[r] = fmaxf(s[0][r], s[1][r]);
        }
        #pragma unroll
        for (int off = 8; off >= 1; off >>= 1) {
            #pragma unroll
            for (int r = 0; r < 8; ++r)
                mx[r] = fmaxf(mx[r], __shfl_xor(mx[r], off, 32));
        }

        // online softmax update
        float alpha[8], rs[8];
        #pragma unroll
        for (int r = 0; r < 8; ++r) {
            const float mnew = fmaxf(mrun[r], mx[r]);
            alpha[r] = __expf(mrun[r] - mnew);
            mrun[r] = mnew;
            const float p0 = __expf(s[0][r] - mnew);
            const float p1 = __expf(s[1][r] - mnew);
            s[0][r] = p0; s[1][r] = p1;
            rs[r] = p0 + p1;
        }
        #pragma unroll
        for (int off = 8; off >= 1; off >>= 1) {
            #pragma unroll
            for (int r = 0; r < 8; ++r)
                rs[r] += __shfl_xor(rs[r], off, 32);
        }
        #pragma unroll
        for (int r = 0; r < 8; ++r) {
            lrun[r] = lrun[r] * alpha[r] + rs[r];
            #pragma unroll
            for (int nt = 0; nt < 4; ++nt) oacc[nt][r] *= alpha[r];
        }

        // P (C-layout) -> LDS -> A fragment
        #pragma unroll
        for (int r = 0; r < 8; ++r) {
            #pragma unroll
            for (int j = 0; j < 2; ++j)
                plds[wv][r + hh * 8][j * 16 + rlo] = f2bf(s[j][r]);
        }
        asm volatile("s_wait_dscnt 0" ::: "memory");   // same-wave LDS RAW across lanes

        const v16bf pa = load_a_frag(&plds[wv][0][0], 32, 0, 0, lane);

        // O += P @ V  (V fragments already in flight / in registers)
        #pragma unroll
        for (int nt = 0; nt < 4; ++nt)
            oacc[nt] = __builtin_amdgcn_wmma_f32_16x16x32_bf16(
                false, pa, false, vb[nt], (short)0, oacc[nt], false, false);
    }

    // normalize and store to [B, T, D] bf16 for the output projection
    #pragma unroll
    for (int nt = 0; nt < 4; ++nt) {
        #pragma unroll
        for (int r = 0; r < 8; ++r) {
            const int t  = qbase + r + hh * 8;
            const int hd = nt * 16 + rlo;
            const float v = oacc[nt][r] / lrun[r];
            Obf[((size_t)b * SEQ + t) * D_MODEL + h * HEAD_DIM + hd] = f2bf(v);
        }
    }
}

// ---------- output projection GEMM: [8192,1024] x [1024,1024] + bias -> f32 ----------

__global__ __launch_bounds__(256)
__attribute__((amdgpu_waves_per_eu(1, 2)))
void out_gemm_kernel(
    const __bf16* __restrict__ A, const __bf16* __restrict__ Wt,
    const float* __restrict__ bias, float* __restrict__ out)
{
    const int lane  = threadIdx.x & 31;
    const int wv    = threadIdx.x >> 5;
    const int nbase = blockIdx.x * 128;
    const int mrow  = blockIdx.y * 128 + wv * 16;

    GEMM_CORE(A, Wt)

    const int rlo = lane & 15, hh = lane >> 4;
    #pragma unroll
    for (int nt = 0; nt < 8; ++nt) {
        const int ng = nbase + nt * 16 + rlo;
        const float bia = bias[ng];
        #pragma unroll
        for (int r = 0; r < 8; ++r) {
            const int mg = mrow + r + hh * 8;
            out[(size_t)mg * D_MODEL + ng] = acc[nt][r] + bia;
        }
    }
}

// ---------- launch ----------

extern "C" void kernel_launch(void* const* d_in, const int* in_sizes, int n_in,
                              void* d_out, int out_size, void* d_ws, size_t ws_size,
                              hipStream_t stream) {
    (void)in_sizes; (void)n_in; (void)out_size; (void)ws_size;

    const float* x    = (const float*)d_in[0];
    const float* Wqkv = (const float*)d_in[1];
    const float* bqkv = (const float*)d_in[2];
    const float* Wout = (const float*)d_in[3];
    const float* bout = (const float*)d_in[4];
    float* out = (float*)d_out;

    char* ws = (char*)d_ws;
    size_t off = 0;
    auto alloc = [&](size_t bytes) {
        char* p = ws + off;
        off += (bytes + 255) & ~(size_t)255;
        return p;
    };

    __bf16* Xbf   = (__bf16*)alloc((size_t)M_TOK * D_MODEL * 2);
    __bf16* WqkvT = (__bf16*)alloc((size_t)3 * D_MODEL * D_MODEL * 2);
    __bf16* WoutT = (__bf16*)alloc((size_t)D_MODEL * D_MODEL * 2);
    __bf16* Qb    = (__bf16*)alloc((size_t)M_TOK * D_MODEL * 2);
    __bf16* Kb    = (__bf16*)alloc((size_t)M_TOK * D_MODEL * 2);
    __bf16* Vtb   = (__bf16*)alloc((size_t)M_TOK * D_MODEL * 2);
    __bf16* Abf   = (__bf16*)alloc((size_t)M_TOK * D_MODEL * 2);

    const int nX = M_TOK * D_MODEL;
    f32_to_bf16_kernel<<<(nX + 255) / 256, 256, 0, stream>>>(x, Xbf, nX);

    const int nWq = 3 * D_MODEL * D_MODEL;
    transpose_bf16_kernel<<<(nWq + 255) / 256, 256, 0, stream>>>(Wqkv, WqkvT, D_MODEL, 3 * D_MODEL);

    const int nWo = D_MODEL * D_MODEL;
    transpose_bf16_kernel<<<(nWo + 255) / 256, 256, 0, stream>>>(Wout, WoutT, D_MODEL, D_MODEL);

    qkv_gemm_kernel<<<dim3(3 * D_MODEL / 128, M_TOK / 128), 256, 0, stream>>>(
        Xbf, WqkvT, bqkv, Qb, Kb, Vtb);

    attn_kernel<<<dim3(BATCH * N_HEADS, SEQ / 128), 256, 0, stream>>>(Qb, Kb, Vtb, Abf);

    out_gemm_kernel<<<dim3(D_MODEL / 128, M_TOK / 128), 256, 0, stream>>>(
        Abf, WoutT, bout, out);
}
